// HierarchicalSparseAttention_3427383902611
// MI455X (gfx1250) — compile-verified, matrix operand order checked by reading, not run on hardware
//
#include <hip/hip_runtime.h>
#include <hip/hip_bf16.h>
#include <math.h>

// ---------------- types ----------------
typedef __hip_bfloat16 hbf16;
typedef __attribute__((ext_vector_type(16))) __bf16 v16bf;
typedef __attribute__((ext_vector_type(8)))  float  v8f;

#define B_     4
#define N_     4096
#define D_     512
#define H_     8
#define NM1_   4095
#define NODES_ 8191
#define LVLS_  12
#define SCALE_ 0.125f   // 1/sqrt(64)
#define NEG_BIG_ (-3.402823466e38f)

union FragAB {
  uint4 u[2];   // 32 bytes
  v16bf v;      // 16 x bf16
};

// ---------------- elementwise helpers ----------------
__global__ void f32_to_bf16_kernel(const float* __restrict__ src,
                                   hbf16* __restrict__ dst, int n) {
  int t = blockIdx.x * blockDim.x + threadIdx.x;
  if (t < n) dst[t] = __float2bfloat16(src[t]);
}

// dst[b, dst_off+p, :] = 0.5*(src[b, src_off+2p, :] + src[b, src_off+2p+1, :])
__global__ void avg_pairs_kernel(const hbf16* __restrict__ src, hbf16* __restrict__ dst,
                                 int P, int src_rpb, int src_off,
                                 int dst_rpb, int dst_off) {
  int t = blockIdx.x * blockDim.x + threadIdx.x;
  int total = B_ * P * D_;
  if (t >= total) return;
  int d = t % D_;
  int p = (t / D_) % P;
  int b = t / (D_ * P);
  size_t s0 = ((size_t)(b * src_rpb + src_off + 2 * p)) * D_ + d;
  float v = 0.5f * (__bfloat162float(src[s0]) + __bfloat162float(src[s0 + D_]));
  dst[((size_t)(b * dst_rpb + dst_off + p)) * D_ + d] = __float2bfloat16(v);
}

// nodes[b, 0:N, :] = x[b, :, :]
__global__ void copy_leaves_kernel(const hbf16* __restrict__ xb,
                                   hbf16* __restrict__ nodes) {
  int t = blockIdx.x * blockDim.x + threadIdx.x;
  const int total = B_ * N_ * D_;
  if (t >= total) return;
  int b = t / (N_ * D_);
  int r = t - b * (N_ * D_);
  nodes[(size_t)b * NODES_ * D_ + r] = xb[t];
}

// ---------------- WMMA GEMM: C[M,512] = A[M,512] * W^T (+bias) ----------------
// Logical A row m maps to physical row (m/a_rpb)*a_bstride + a_off + (m%a_rpb).
// Logical C row m maps to physical row (m/o_rpb)*o_bstride + o_off + (m%o_rpb).
// Each wave computes a 32x64 output tile: 2 A fragments, 4 B fragments,
// 8 f32 accumulators, 8 x v_wmma_f32_16x16x32_bf16 per K-step (K stepped by 32),
// with double-buffered fragment loads so VMEM overlaps the matrix pipe.
__launch_bounds__(128)
__global__ void gemm_bf16_wmma_kernel(
    const hbf16* __restrict__ A, const hbf16* __restrict__ W,
    const float* __restrict__ bias,
    float* __restrict__ Cf, hbf16* __restrict__ Cb,
    int M, int a_rpb, int a_bstride, int a_off,
    int o_rpb, int o_bstride, int o_off) {
  int lane  = threadIdx.x & 31;
  int wave  = threadIdx.x >> 5;
  int mtile = blockIdx.x * 4 + wave;          // 32 rows per wave
  if (mtile * 32 >= M) return;                // uniform per wave
  int n0  = blockIdx.y * 64;
  int l16 = lane & 15;
  int kb  = (lane >> 4) * 8;                  // K-half split across lane groups

  // A fragment rows for this lane; clamp out-of-range rows to M-1
  // (their products land in C rows >= M which are never stored).
  int mlo = mtile * 32 + l16;
  int mhi = mlo + 16;
  mlo = (mlo < M) ? mlo : (M - 1);
  mhi = (mhi < M) ? mhi : (M - 1);
  int ablo = mlo / a_rpb, ajlo = mlo - ablo * a_rpb;
  int abhi = mhi / a_rpb, ajhi = mhi - abhi * a_rpb;
  const hbf16* aptr_lo = A + ((size_t)ablo * a_bstride + a_off + ajlo) * D_;
  const hbf16* aptr_hi = A + ((size_t)abhi * a_bstride + a_off + ajhi) * D_;

  // B fragments: B[k][n] = W[n][k]; row n contiguous in k
  const hbf16* wp0 = W + (size_t)(n0 + l16) * D_;

  v8f acc0 = {0,0,0,0,0,0,0,0};
  v8f acc1 = {0,0,0,0,0,0,0,0};
  v8f acc2 = {0,0,0,0,0,0,0,0};
  v8f acc3 = {0,0,0,0,0,0,0,0};
  v8f acc4 = {0,0,0,0,0,0,0,0};
  v8f acc5 = {0,0,0,0,0,0,0,0};
  v8f acc6 = {0,0,0,0,0,0,0,0};
  v8f acc7 = {0,0,0,0,0,0,0,0};

  FragAB alo[2], ahi[2], bfrag[2][4];

  auto load_step = [&](int k0, int s) {
    alo[s].u[0] = *(const uint4*)(aptr_lo + k0 + kb);
    alo[s].u[1] = *(const uint4*)(aptr_lo + k0 + 16 + kb);
    ahi[s].u[0] = *(const uint4*)(aptr_hi + k0 + kb);
    ahi[s].u[1] = *(const uint4*)(aptr_hi + k0 + 16 + kb);
    #pragma unroll
    for (int t = 0; t < 4; ++t) {
      bfrag[s][t].u[0] = *(const uint4*)(wp0 + t * 16 * D_ + k0 + kb);
      bfrag[s][t].u[1] = *(const uint4*)(wp0 + t * 16 * D_ + k0 + 16 + kb);
    }
  };
  auto do_wmma = [&](int s) {
    acc0 = __builtin_amdgcn_wmma_f32_16x16x32_bf16(false, alo[s].v, false, bfrag[s][0].v, (short)0, acc0, false, false);
    acc1 = __builtin_amdgcn_wmma_f32_16x16x32_bf16(false, alo[s].v, false, bfrag[s][1].v, (short)0, acc1, false, false);
    acc2 = __builtin_amdgcn_wmma_f32_16x16x32_bf16(false, alo[s].v, false, bfrag[s][2].v, (short)0, acc2, false, false);
    acc3 = __builtin_amdgcn_wmma_f32_16x16x32_bf16(false, alo[s].v, false, bfrag[s][3].v, (short)0, acc3, false, false);
    acc4 = __builtin_amdgcn_wmma_f32_16x16x32_bf16(false, ahi[s].v, false, bfrag[s][0].v, (short)0, acc4, false, false);
    acc5 = __builtin_amdgcn_wmma_f32_16x16x32_bf16(false, ahi[s].v, false, bfrag[s][1].v, (short)0, acc5, false, false);
    acc6 = __builtin_amdgcn_wmma_f32_16x16x32_bf16(false, ahi[s].v, false, bfrag[s][2].v, (short)0, acc6, false, false);
    acc7 = __builtin_amdgcn_wmma_f32_16x16x32_bf16(false, ahi[s].v, false, bfrag[s][3].v, (short)0, acc7, false, false);
  };

  // software pipeline: preload step 0, load step i+1 while computing step i
  load_step(0, 0);
  int s = 0;
  #pragma unroll
  for (int k0 = 0; k0 < D_ - 32; k0 += 32) {
    load_step(k0 + 32, s ^ 1);
    do_wmma(s);
    s ^= 1;
  }
  do_wmma(s);

  // C layout: lane l holds N = n0 + 16*t + (l&15); rows M = v + 8*(l>>4)
  int mbase = mtile * 32 + 8 * (lane >> 4);
  #pragma unroll
  for (int blk = 0; blk < 2; ++blk) {
    #pragma unroll
    for (int v = 0; v < 8; ++v) {
      int mr = mbase + blk * 16 + v;
      if (mr >= M) continue;
      int ob = mr / o_rpb;
      int oj = mr - ob * o_rpb;
      size_t orow = ((size_t)ob * o_bstride + o_off + oj) * D_;
      float vv[4];
      if (blk == 0) { vv[0] = acc0[v]; vv[1] = acc1[v]; vv[2] = acc2[v]; vv[3] = acc3[v]; }
      else          { vv[0] = acc4[v]; vv[1] = acc5[v]; vv[2] = acc6[v]; vv[3] = acc7[v]; }
      #pragma unroll
      for (int t = 0; t < 4; ++t) {
        int nc = n0 + 16 * t + l16;
        float val = vv[t];
        if (bias) val += bias[nc];
        if (Cf) Cf[orow + nc] = val;
        if (Cb) Cb[orow + nc] = __float2bfloat16(val);
      }
    }
  }
}

// ---------------- attention helpers ----------------
__device__ __forceinline__ float wave_dot(const hbf16* __restrict__ kp,
                                          float q0, float q1, int d0) {
  float p = q0 * __bfloat162float(kp[d0]) + q1 * __bfloat162float(kp[d0 + 32]);
  #pragma unroll
  for (int s = 16; s > 0; s >>= 1) p += __shfl_xor(p, s, 32);
  return p;
}

// One wave per (b, p, h): parent attends over its two children.
__global__ void tree_attn_kernel(const hbf16* __restrict__ Q, const hbf16* __restrict__ K,
                                 const hbf16* __restrict__ V, hbf16* __restrict__ newlev,
                                 int P, int off) {
  int gw   = (blockIdx.x * blockDim.x + threadIdx.x) >> 5;
  int lane = threadIdx.x & 31;
  int total = B_ * P * H_;
  if (gw >= total) return;
  int h = gw % H_;
  int p = (gw / H_) % P;
  int b = gw / (H_ * P);
  int d0 = h * 64 + lane;

  size_t qrow  = ((size_t)(b * NM1_ + off + p)) * D_;
  float q0 = __bfloat162float(Q[qrow + d0]);
  float q1 = __bfloat162float(Q[qrow + d0 + 32]);

  size_t krow = ((size_t)(b * 2 * P + 2 * p)) * D_;
  float l0 = wave_dot(K + krow,       q0, q1, d0) * SCALE_;
  float l1 = wave_dot(K + krow + D_,  q0, q1, d0) * SCALE_;
  float m  = fmaxf(l0, l1);
  float e0 = __expf(l0 - m), e1 = __expf(l1 - m);
  float inv = 1.0f / (e0 + e1 + 1e-9f);
  float w0 = e0 * inv, w1 = e1 * inv;

  float o0 = w0 * __bfloat162float(V[krow + d0])      + w1 * __bfloat162float(V[krow + D_ + d0]);
  float o1 = w0 * __bfloat162float(V[krow + d0 + 32]) + w1 * __bfloat162float(V[krow + D_ + d0 + 32]);
  newlev[qrow + d0]      = __float2bfloat16(o0);
  newlev[qrow + d0 + 32] = __float2bfloat16(o1);
}

// One wave per (b, n, h): leaf attends to self + allowed sibling path nodes.
__global__ void leaf_attn_kernel(const hbf16* __restrict__ Qx, const hbf16* __restrict__ K,
                                 const hbf16* __restrict__ V, hbf16* __restrict__ out) {
  int gw   = (blockIdx.x * blockDim.x + threadIdx.x) >> 5;
  int lane = threadIdx.x & 31;
  const int total = B_ * N_ * H_;
  if (gw >= total) return;
  int h = gw % H_;
  int n = (gw / H_) % N_;
  int b = gw / (H_ * N_);
  int d0 = h * 64 + lane;

  size_t qrow = ((size_t)(b * N_ + n)) * D_;
  float q0 = __bfloat162float(Qx[qrow + d0]);
  float q1 = __bfloat162float(Qx[qrow + d0 + 32]);

  // Build sibling-path table (matches _build_table in the reference)
  bool allow[13];
  int  rows[13];
  allow[0] = true;
  rows[0]  = b * NODES_ + n;
  int ncur = n;
  bool alive = true;
  #pragma unroll
  for (int lvl = 0; lvl < LVLS_; ++lvl) {
    int pair, nn;
    if (lvl == 0) { pair = ncur;              nn = ncur ^ 1; }
    else          { pair = (ncur >> 1) + N_;  nn = pair ^ 1; }
    bool ok = alive && (nn <= 2 * N_ - 3);
    allow[lvl + 1] = ok && !(pair < nn);
    rows[lvl + 1]  = b * NODES_ + (ok ? nn : 0);
    if (ok) ncur = nn;
    alive = ok;
  }

  float lg[13];
  #pragma unroll
  for (int i = 0; i < 13; ++i)
    lg[i] = allow[i] ? wave_dot(K + (size_t)rows[i] * D_, q0, q1, d0) * SCALE_
                     : NEG_BIG_;
  float m = lg[0];
  #pragma unroll
  for (int i = 1; i < 13; ++i) m = fmaxf(m, lg[i]);
  float w[13];
  float s = 0.f;
  #pragma unroll
  for (int i = 0; i < 13; ++i) {
    w[i] = allow[i] ? __expf(lg[i] - m) : 0.f;
    s += w[i];
  }
  float inv = 1.0f / s;
  float o0 = 0.f, o1 = 0.f;
  #pragma unroll
  for (int i = 0; i < 13; ++i) {
    if (!allow[i]) continue;
    float wi = w[i] * inv;
    const hbf16* vp = V + (size_t)rows[i] * D_;
    o0 += wi * __bfloat162float(vp[d0]);
    o1 += wi * __bfloat162float(vp[d0 + 32]);
  }
  out[qrow + d0]      = __float2bfloat16(o0);
  out[qrow + d0 + 32] = __float2bfloat16(o1);
}

// ---------------- host orchestration ----------------
extern "C" void kernel_launch(void* const* d_in, const int* in_sizes, int n_in,
                              void* d_out, int out_size, void* d_ws, size_t ws_size,
                              hipStream_t stream) {
  (void)in_sizes; (void)n_in; (void)out_size; (void)ws_size;

  const float* x    = (const float*)d_in[0];
  const float* Wf[8] = {
    (const float*)d_in[1], (const float*)d_in[2], (const float*)d_in[3],
    (const float*)d_in[4], (const float*)d_in[5], (const float*)d_in[6],
    (const float*)d_in[7], (const float*)d_in[8] };
  const float* bo_y = (const float*)d_in[9];
  const float* bo_x = (const float*)d_in[10];

  // workspace carving (256B aligned slots)
  char* ws = (char*)d_ws;
  size_t o = 0;
  auto carve = [&](size_t bytes) -> char* {
    char* p = ws + o;
    o = (o + bytes + 255) & ~(size_t)255;
    return p;
  };
  hbf16* xb    = (hbf16*)carve((size_t)B_ * N_    * D_ * 2);  // bf16 x
  hbf16* yb    = (hbf16*)carve((size_t)B_ * NM1_  * D_ * 2);  // pooled hierarchy
  hbf16* Wb    = (hbf16*)carve((size_t)8  * D_    * D_ * 2);  // 8 bf16 weights
  hbf16* Qb    = (hbf16*)carve((size_t)B_ * N_    * D_ * 2);  // Q_all, later Qx
  hbf16* Kb    = (hbf16*)carve((size_t)B_ * 8192  * D_ * 2);  // level K, later node K
  hbf16* Vb    = (hbf16*)carve((size_t)B_ * 8192  * D_ * 2);  // level V, later node V
  hbf16* nl    = (hbf16*)carve((size_t)B_ * NM1_  * D_ * 2);  // updated levels
  hbf16* nodes = (hbf16*)carve((size_t)B_ * NODES_* D_ * 2);  // [x ; Y]
  hbf16* ao    = (hbf16*)carve((size_t)B_ * N_    * D_ * 2);  // attention out

  const size_t DD = (size_t)D_ * D_;
  auto gemm = [&](const hbf16* A, const hbf16* Wm, const float* bias,
                  float* Cf, hbf16* Cb, int M,
                  int arpb, int abstride, int aoff,
                  int orpb, int obstride, int ooff) {
    dim3 g((M + 127) / 128, D_ / 64, 1);   // 4 waves x 32 rows per block
    gemm_bf16_wmma_kernel<<<g, 128, 0, stream>>>(A, Wm, bias, Cf, Cb, M,
                                                 arpb, abstride, aoff,
                                                 orpb, obstride, ooff);
  };

  // 1) convert x and weights to bf16
  const int nx = B_ * N_ * D_;
  f32_to_bf16_kernel<<<(nx + 255) / 256, 256, 0, stream>>>(x, xb, nx);
  for (int i = 0; i < 8; ++i)
    f32_to_bf16_kernel<<<((int)DD + 255) / 256, 256, 0, stream>>>(Wf[i], Wb + i * DD, (int)DD);

  // 2) build pooled hierarchy y_in (level l at offset N - (N>>l))
  for (int lvl = 0; lvl < LVLS_; ++lvl) {
    int P   = N_ >> (lvl + 1);
    int off = N_ - (N_ >> lvl);
    int total = B_ * P * D_;
    if (lvl == 0)
      avg_pairs_kernel<<<(total + 255) / 256, 256, 0, stream>>>(xb, yb, P, N_, 0, NM1_, 0);
    else {
      int poff = N_ - (N_ >> (lvl - 1));
      avg_pairs_kernel<<<(total + 255) / 256, 256, 0, stream>>>(yb, yb, P, NM1_, poff, NM1_, off);
    }
  }

  // 3) Q_all = y_in @ Wq_y^T
  gemm(yb, Wb + 0 * DD, nullptr, nullptr, Qb, B_ * NM1_,
       B_ * NM1_, 0, 0, B_ * NM1_, 0, 0);

  // 4) bottom-up tree attention
  for (int lvl = 0; lvl < LVLS_; ++lvl) {
    int P   = N_ >> (lvl + 1);
    int off = N_ - (N_ >> lvl);
    int M   = B_ * 2 * P;
    const hbf16* Aprev = (lvl == 0) ? xb : nl;
    int abstride = (lvl == 0) ? N_ : NM1_;
    int aoff     = (lvl == 0) ? 0  : (N_ - (N_ >> (lvl - 1)));
    gemm(Aprev, Wb + 1 * DD, nullptr, nullptr, Kb, M, 2 * P, abstride, aoff, M, 0, 0);
    gemm(Aprev, Wb + 2 * DD, nullptr, nullptr, Vb, M, 2 * P, abstride, aoff, M, 0, 0);
    int waves = B_ * P * H_;
    tree_attn_kernel<<<(waves * 32 + 255) / 256, 256, 0, stream>>>(Qb, Kb, Vb, nl, P, off);
  }

  // 5) Y = newlev @ Wo_y^T + bo_y, scattered straight into nodes[:, N:, :]
  gemm(nl, Wb + 3 * DD, bo_y, nullptr, nodes, B_ * NM1_,
       B_ * NM1_, 0, 0, NM1_, NODES_, N_);
  copy_leaves_kernel<<<(nx + 255) / 256, 256, 0, stream>>>(xb, nodes);

  // 6) cross projections
  gemm(xb,    Wb + 4 * DD, nullptr, nullptr, Qb, B_ * N_,
       B_ * N_, 0, 0, B_ * N_, 0, 0);
  gemm(nodes, Wb + 5 * DD, nullptr, nullptr, Kb, B_ * NODES_,
       B_ * NODES_, 0, 0, B_ * NODES_, 0, 0);
  gemm(nodes, Wb + 6 * DD, nullptr, nullptr, Vb, B_ * NODES_,
       B_ * NODES_, 0, 0, B_ * NODES_, 0, 0);

  // 7) leaf attention over self + sibling path
  int lwaves = B_ * N_ * H_;
  leaf_attn_kernel<<<(lwaves * 32 + 255) / 256, 256, 0, stream>>>(Qb, Kb, Vb, ao);

  // 8) final projection to f32 output
  gemm(ao, Wb + 7 * DD, bo_x, (float*)d_out, nullptr, B_ * N_,
       B_ * N_, 0, 0, B_ * N_, 0, 0);
}
